// CustomLSTM_84241488544403
// MI455X (gfx1250) — matrix operand order repaired and checked by reading, not compiled, not run on hardware
//
#include <hip/hip_runtime.h>

// ---------------------------------------------------------------------------
// CDNA5 (gfx1250) persistent-LSTM, barrier-latency-hiding version:
//   - persistent kernel runs all 512 timesteps (device-wide atomic barrier)
//   - W (1536x4096 bf16, column-major) resident in LDS, filled once via
//     global_load_async_to_lds_b128 / s_wait_asynccnt (ASYNCcnt path)
//   - gates via v_wmma_f32_16x16x32_bf16, B-operands streamed from LDS
//   - cell state c entirely in registers; h ping-pongs in global bf16
//   - x-projection of step t+1 and hseq stores overlap the step-t barrier
// ---------------------------------------------------------------------------

typedef __attribute__((ext_vector_type(8)))  float  v8f;
typedef __attribute__((ext_vector_type(16))) __bf16 v16bf;
typedef __attribute__((ext_vector_type(8)))  __bf16 v8bf;

#define BS    32
#define TSEQ  512
#define INSZ  512
#define HID   1024
#define KTOT  (INSZ + HID)        // 1536
#define NTOT  (4 * HID)           // 4096
#define NWG   64                  // persistent workgroups, one n_tile each
#define WGSZ  64                  // 2 waves: m_tile 0/1
#define LCOL  (KTOT + 8)          // padded LDS column stride (elems) -> conflict-free
#define LDSB  (64 * LCOL * 2)     // 197632 B dynamic LDS (<= 320KB WGP limit)

// ---------------- one-time prep kernels ----------------

__global__ void k_cvt_x(const float* __restrict__ x, __bf16* __restrict__ xbf, int n) {
    int i = blockIdx.x * blockDim.x + threadIdx.x;
    if (i < n) xbf[i] = (__bf16)x[i];
}

// Column-major bf16 weights: wcm[col*1536 + k], col = gate*1024 + n.
__global__ void k_build_w(const float* __restrict__ Uf, const float* __restrict__ Vf,
                          const float* __restrict__ Ui, const float* __restrict__ Vi,
                          const float* __restrict__ Uo, const float* __restrict__ Vo,
                          const float* __restrict__ Ug, const float* __restrict__ Vg,
                          __bf16* __restrict__ wcm) {
    int idx = blockIdx.x * blockDim.x + threadIdx.x;   // [0, 4096*1536)
    int col = idx / KTOT;
    int k   = idx - col * KTOT;
    int g   = col >> 10;
    int n   = col & (HID - 1);
    const float* U = (g == 0) ? Uf : (g == 1) ? Ui : (g == 2) ? Uo : Ug;
    const float* V = (g == 0) ? Vf : (g == 1) ? Vi : (g == 2) ? Vo : Vg;
    float v = (k < INSZ) ? U[k * HID + n] : V[(k - INSZ) * HID + n];
    wcm[idx] = (__bf16)v;
}

__global__ void k_init(__bf16* __restrict__ h0, __bf16* __restrict__ h1,
                       unsigned* __restrict__ ctr, int n) {
    int i = blockIdx.x * blockDim.x + threadIdx.x;
    if (i == 0) *ctr = 0u;
    if (i < n) { h0[i] = (__bf16)0.0f; h1[i] = (__bf16)0.0f; }
}

// ---------------- persistent LSTM kernel ----------------

__device__ __forceinline__ float sigmoid_f(float x) {
    return 1.0f / (1.0f + __expf(-x));
}
__device__ __forceinline__ float tanh_f(float x) {
    return 2.0f / (1.0f + __expf(-2.0f * x)) - 1.0f;
}

union AV { v16bf v; v8bf h[2]; };

// Accumulate NKT k-tiles: A from a_ptr (row-major, khalf already applied),
// B from the 4 per-gate LDS columns at element offset koff.
template <int NKT>
__device__ __forceinline__ void gemm_part(const __bf16* __restrict__ a_ptr,
                                          const __bf16* __restrict__ wl0,
                                          const __bf16* __restrict__ wl1,
                                          const __bf16* __restrict__ wl2,
                                          const __bf16* __restrict__ wl3,
                                          int koff,
                                          v8f& acc0, v8f& acc1, v8f& acc2, v8f& acc3)
{
#pragma unroll 4
    for (int kt = 0; kt < NKT; ++kt) {
        const int kk = kt * 32;
        const int ko = koff + kk;
        AV a, b0, b1, b2, b3;
        a.h[0]  = *(const v8bf*)(a_ptr + kk);
        a.h[1]  = *(const v8bf*)(a_ptr + kk + 8);
        b0.h[0] = *(const v8bf*)(wl0 + ko);  b0.h[1] = *(const v8bf*)(wl0 + ko + 8);
        b1.h[0] = *(const v8bf*)(wl1 + ko);  b1.h[1] = *(const v8bf*)(wl1 + ko + 8);
        b2.h[0] = *(const v8bf*)(wl2 + ko);  b2.h[1] = *(const v8bf*)(wl2 + ko + 8);
        b3.h[0] = *(const v8bf*)(wl3 + ko);  b3.h[1] = *(const v8bf*)(wl3 + ko + 8);
        acc0 = __builtin_amdgcn_wmma_f32_16x16x32_bf16(false, a.v, false, b0.v, (short)0, acc0, false, false);
        acc1 = __builtin_amdgcn_wmma_f32_16x16x32_bf16(false, a.v, false, b1.v, (short)0, acc1, false, false);
        acc2 = __builtin_amdgcn_wmma_f32_16x16x32_bf16(false, a.v, false, b2.v, (short)0, acc2, false, false);
        acc3 = __builtin_amdgcn_wmma_f32_16x16x32_bf16(false, a.v, false, b3.v, (short)0, acc3, false, false);
    }
}

__global__ void __launch_bounds__(WGSZ)
k_lstm(const __bf16* __restrict__ xbf, const __bf16* __restrict__ wcm,
       __bf16* __restrict__ hg0, __bf16* __restrict__ hg1,
       const float* __restrict__ bF, const float* __restrict__ bI,
       const float* __restrict__ bO, const float* __restrict__ bG,
       float* __restrict__ hseq, float* __restrict__ hlast,
       unsigned* __restrict__ bar)
{
    extern __shared__ __align__(16) __bf16 smem[];   // [64 cols][LCOL elems]
    const int tid    = threadIdx.x;
    const int lane   = tid & 31;
    const int m_tile = tid >> 5;                     // 0/1
    const int n_tile = blockIdx.x;                   // 0..63

    // ---- prologue: async-fill this block's W slice into LDS ----
    {
        const int      cl   = tid;                   // one LDS column per thread
        const int      g    = cl >> 4, nl = cl & 15;
        const unsigned gcol = (unsigned)(g * HID + n_tile * 16 + nl);
        const unsigned long long sbase = (unsigned long long)(uintptr_t)wcm;
        unsigned voff = gcol * (unsigned)(KTOT * 2);
        unsigned loff = (unsigned)(uintptr_t)(void*)smem
                      + (unsigned)cl * (unsigned)(LCOL * 2);
#pragma unroll 8
        for (int p = 0; p < (KTOT * 2) / 16; ++p) {  // 192 x 16B chunks
            asm volatile("global_load_async_to_lds_b128 %0, %1, %2"
                         :: "v"(loff), "v"(voff), "s"(sbase) : "memory");
            voff += 16u; loff += 16u;
        }
        asm volatile("s_wait_asynccnt 0" ::: "memory");
    }
    __syncthreads();

    // Fragment addressing (this code's contiguous k order, same in A and B):
    //  A 16x32 bf16: row m = lane&15, 16 consecutive K at (lane>>4)*16 (2x16B)
    //  B 32x16 bf16: col n = lane&15, 16 consecutive K at (lane>>4)*16 (2x16B)
    //  D 16x16 f32 : m = (lane>=16?8:0)+j, n = lane&15
    const int am    = m_tile * 16 + (lane & 15);
    const int khalf = (lane >> 4) * 16;
    const int nc    = n_tile * 16 + (lane & 15);
    const int kb    = (lane >> 4) * 16;

    const __bf16* wl0 = smem + (size_t)(0 * 16 + (lane & 15)) * LCOL + kb;
    const __bf16* wl1 = smem + (size_t)(1 * 16 + (lane & 15)) * LCOL + kb;
    const __bf16* wl2 = smem + (size_t)(2 * 16 + (lane & 15)) * LCOL + kb;
    const __bf16* wl3 = smem + (size_t)(3 * 16 + (lane & 15)) * LCOL + kb;

    const float biasF = bF[nc], biasI = bI[nc], biasO = bO[nc], biasG = bG[nc];
    const int   mbase = m_tile * 16 + ((lane >> 4) << 3);
    const __bf16* xrow = xbf + (size_t)am * (TSEQ * INSZ) + khalf;

    float cr[8] = {0.f, 0.f, 0.f, 0.f, 0.f, 0.f, 0.f, 0.f};   // register cell state

    // Pre-seed accumulators with the x-projection of step 0.
    v8f xacc0 = {}, xacc1 = {}, xacc2 = {}, xacc3 = {};
    gemm_part<INSZ / 32>(xrow, wl0, wl1, wl2, wl3, 0, xacc0, xacc1, xacc2, xacc3);

    for (int t = 0; t < TSEQ; ++t) {
        const __bf16* hin  = (t & 1) ? hg1 : hg0;
        __bf16*       hout = (t & 1) ? hg0 : hg1;
        const __bf16* ha   = hin + am * HID + khalf;

        // ---- serialized part: h-projection on top of pre-seeded x accs ----
        v8f acc0 = xacc0, acc1 = xacc1, acc2 = xacc2, acc3 = xacc3;
        gemm_part<HID / 32>(ha, wl0, wl1, wl2, wl3, INSZ, acc0, acc1, acc2, acc3);

        // ---- fused gate epilogue; store ONLY the h ping-pong (L2) here ----
        float hn[8];
#pragma unroll
        for (int j = 0; j < 8; ++j) {
            float fg = sigmoid_f(acc0[j] + biasF);
            float ig = sigmoid_f(acc1[j] + biasI);
            float og = sigmoid_f(acc2[j] + biasO);
            float gg = tanh_f   (acc3[j] + biasG);
            float cn = fg * cr[j] + ig * gg;
            hn[j]    = og * tanh_f(cn);
            cr[j]    = cn;
            hout[(mbase + j) * HID + nc] = (__bf16)hn[j];
        }

        // Release h writes and arrive at the device barrier ASAP.
        __threadfence();
        __syncthreads();
        if (tid == 0)
            __hip_atomic_fetch_add(bar, 1u, __ATOMIC_ACQ_REL, __HIP_MEMORY_SCOPE_AGENT);

        // ---- overlap window (independent of other blocks' progress) ----
        //  1) hseq/hlast output stores (HBM latency hidden behind barrier)
#pragma unroll
        for (int j = 0; j < 8; ++j) {
            const int m = mbase + j;
            hseq[((size_t)m * TSEQ + t) * HID + nc] = hn[j];
            if (t == TSEQ - 1) hlast[m * HID + nc] = hn[j];
        }
        //  2) x-projection of step t+1 (does not depend on h_t)
        xacc0 = (v8f){}; xacc1 = (v8f){}; xacc2 = (v8f){}; xacc3 = (v8f){};
        if (t + 1 < TSEQ) {
            const __bf16* xa = xrow + (size_t)(t + 1) * INSZ;
            if (t + 2 < TSEQ) __builtin_prefetch(xa + INSZ, 0, 0);
            gemm_part<INSZ / 32>(xa, wl0, wl1, wl2, wl3, 0, xacc0, xacc1, xacc2, xacc3);
        }

        // ---- wait for all blocks, then next step may read h_t ----
        if (tid == 0) {
            const unsigned target = (unsigned)(t + 1) * NWG;
            while (__hip_atomic_load(bar, __ATOMIC_ACQUIRE, __HIP_MEMORY_SCOPE_AGENT) < target)
                __builtin_amdgcn_s_sleep(1);
        }
        __syncthreads();
    }
}

// ---------------- host launch ----------------

extern "C" void kernel_launch(void* const* d_in, const int* in_sizes, int n_in,
                              void* d_out, int out_size, void* d_ws, size_t ws_size,
                              hipStream_t stream)
{
    const float* x  = (const float*)d_in[0];
    const float* Uf = (const float*)d_in[1];
    const float* Vf = (const float*)d_in[2];
    const float* bF = (const float*)d_in[3];
    const float* Ui = (const float*)d_in[4];
    const float* Vi = (const float*)d_in[5];
    const float* bI = (const float*)d_in[6];
    const float* Uo = (const float*)d_in[7];
    const float* Vo = (const float*)d_in[8];
    const float* bO = (const float*)d_in[9];
    const float* Ug = (const float*)d_in[10];
    const float* Vg = (const float*)d_in[11];
    const float* bG = (const float*)d_in[12];
    (void)in_sizes; (void)n_in; (void)out_size; (void)ws_size;

    char* ws = (char*)d_ws;
    size_t off = 0;
    __bf16*   wcm = (__bf16*)(ws + off);   off += (size_t)NTOT * KTOT * sizeof(__bf16);
    __bf16*   xbf = (__bf16*)(ws + off);   off += (size_t)BS * TSEQ * INSZ * sizeof(__bf16);
    __bf16*   hg0 = (__bf16*)(ws + off);   off += (size_t)BS * HID * sizeof(__bf16);
    __bf16*   hg1 = (__bf16*)(ws + off);   off += (size_t)BS * HID * sizeof(__bf16);
    unsigned* bar = (unsigned*)(ws + off); off += 256;

    float* hlast = (float*)d_out;               // [32, 1024]
    float* hseq  = hlast + (size_t)BS * HID;    // [32, 512, 1024]

    // Allow >64KB dynamic LDS for the persistent kernel (idempotent host config).
    (void)hipFuncSetAttribute((const void*)k_lstm,
                              hipFuncAttributeMaxDynamicSharedMemorySize, LDSB);

    { int n = BS * TSEQ * INSZ;
      k_cvt_x<<<(n + 255) / 256, 256, 0, stream>>>(x, xbf, n); }
    { int n = NTOT * KTOT;
      k_build_w<<<(n + 255) / 256, 256, 0, stream>>>(Uf, Vf, Ui, Vi, Uo, Vo, Ug, Vg, wcm); }
    { int n = BS * HID;
      k_init<<<(n + 255) / 256, 256, 0, stream>>>(hg0, hg1, bar, n); }

    k_lstm<<<NWG, WGSZ, LDSB, stream>>>(xbf, wcm, hg0, hg1,
                                        bF, bI, bO, bG, hseq, hlast, bar);
}
